// MultiLinearAttention_37572373905577
// MI455X (gfx1250) — matrix operand order repaired and checked by reading, not compiled, not run on hardware
//
#include <hip/hip_runtime.h>

// ---------------------------------------------------------------------------
// MultiLinearAttention forward for MI455X (gfx1250, wave32, WMMA)
//
// Sizes (fixed by the reference): B=2, L=2048, D_MODEL=512, H=8, d=64, HB=16
// Outputs: out (2*2048*512 f32) then attn (16*2048*2048 f32), concatenated.
//
// Bandwidth-bound: attn output write = 268 MB dominates (~12 us at 23.3 TB/s).
// All GEMM operands (<= 8 MB each) stay L2-resident (192 MB L2), so GEMMs
// read f32 directly from global, convert to f16 in-register, and run on the
// v_wmma_f32_16x16x32_f16 tensor path with f32 accumulation.
// ---------------------------------------------------------------------------

typedef __attribute__((ext_vector_type(16))) _Float16 v16h;
typedef __attribute__((ext_vector_type(8)))  float    v8f;
typedef __attribute__((ext_vector_type(4)))  float    v4f;

#define DM   512
#define LSEQ 2048
#define NB   2
#define NH   8
#define DK   64

// ---------------- fragment loaders (f32 source -> f16 fragment) -------------
// A fragment (16x32, MxK), lane L: row = L%16, g = L/16.
//   halfs[0..7]  = A[row][kb + g*16 + p]
//   halfs[8..15] = A[row][kb + g*16 + 8 + p]
__device__ inline v16h load_a_frag(const float* __restrict__ row, int kb, int g) {
  const float* p0 = row + kb + g * 16;
  const float* p1 = row + kb + g * 16 + 8;
  v4f a0 = *(const v4f*)(p0);
  v4f a1 = *(const v4f*)(p0 + 4);
  v4f a2 = *(const v4f*)(p1);
  v4f a3 = *(const v4f*)(p1 + 4);
  v16h f;
#pragma unroll
  for (int i = 0; i < 4; ++i) {
    f[i]      = (_Float16)a0[i];
    f[4 + i]  = (_Float16)a1[i];
    f[8 + i]  = (_Float16)a2[i];
    f[12 + i] = (_Float16)a3[i];
  }
  return f;
}

// B fragment (32x16, KxN) from BT stored N-major (row n holds K contiguous):
// lane L: col n = L%16, g = L/16; halfs[p] = BT[n][kb + g*16 + p]
__device__ inline v16h load_b_frag(const float* __restrict__ rowBT, int kb, int g) {
  const float* p = rowBT + kb + g * 16;
  v16h f;
#pragma unroll
  for (int q = 0; q < 4; ++q) {
    v4f u = *(const v4f*)(p + 4 * q);
#pragma unroll
    for (int i = 0; i < 4; ++i) f[4 * q + i] = (_Float16)u[i];
  }
  return f;
}

// ---------------- generic batched GEMM:  C = A * BT^T (+bias)(+residual) ----
// A: M x K (lda), BT: N x K (ldb), C: M x N (ldc). Block = 256 thr = 8 waves.
// Block tile 128x128; wave tile 32x64 (2 M-subtiles x 4 N-subtiles).
// Batch z decomposes as z = h*nbInner + b with independent strides.
// HAS_BIAS / HAS_RES are compile-time so the epilogue is branch-free
// (critical for the attn launch: 64 stores/lane, store-bandwidth bound).
template <bool HAS_BIAS, bool HAS_RES>
__global__ __launch_bounds__(256) void gemm_wmma_kernel(
    const float* __restrict__ A, const float* __restrict__ BT,
    const float* __restrict__ bias, const float* __restrict__ residual,
    float* __restrict__ C, int lda, int ldb, int ldc, int K,
    long long sAh, long long sAb, long long sBh, long long sBb,
    long long sCh, long long sCb, int nbInner) {
  const int z = blockIdx.z;
  const int hz = z / nbInner;
  const int bz = z % nbInner;
  A += hz * sAh + bz * sAb;
  BT += hz * sBh + bz * sBb;
  C += hz * sCh + bz * sCb;

  const int tid = threadIdx.x;
  const int wave = tid >> 5;
  const int lane = tid & 31;
  const int g = lane >> 4;   // K-half / row-half select
  const int c = lane & 15;   // A row within subtile == B col within subtile

  const int m0 = blockIdx.y * 128 + (wave >> 1) * 32;
  const int n0 = blockIdx.x * 128 + (wave & 1) * 64;

  v8f acc[2][4] = {};

  for (int kb = 0; kb < K; kb += 32) {
    v16h af[2];
#pragma unroll
    for (int i = 0; i < 2; ++i)
      af[i] = load_a_frag(A + (size_t)(m0 + i * 16 + c) * lda, kb, g);
    v16h bf[4];
#pragma unroll
    for (int j = 0; j < 4; ++j)
      bf[j] = load_b_frag(BT + (size_t)(n0 + j * 16 + c) * ldb, kb, g);
#pragma unroll
    for (int i = 0; i < 2; ++i)
#pragma unroll
      for (int j = 0; j < 4; ++j)
        acc[i][j] = __builtin_amdgcn_wmma_f32_16x16x32_f16(
            false, af[i], false, bf[j], (short)0, acc[i][j], false, false);
  }

  // C/D layout: VGPR r -> row = m0 + i*16 + g*8 + r, col = n0 + j*16 + c
#pragma unroll
  for (int i = 0; i < 2; ++i) {
#pragma unroll
    for (int j = 0; j < 4; ++j) {
      const int col = n0 + j * 16 + c;
      float bv = 0.0f;
      if (HAS_BIAS) bv = bias[col];
#pragma unroll
      for (int r = 0; r < 8; ++r) {
        const int row = m0 + i * 16 + g * 8 + r;
        float val = acc[i][j][r];
        if (HAS_BIAS) val += bv;
        if (HAS_RES) val += residual[(size_t)row * ldc + col];
        C[(size_t)row * ldc + col] = val;
      }
    }
  }
}

// ---------------- weight transpose: WT[n][k] = W[k][n] (512x512) ------------
__global__ __launch_bounds__(256) void transpose_kernel(
    const float* __restrict__ W, float* __restrict__ WT) {
  int idx = blockIdx.x * 256 + threadIdx.x;  // 262144 total
  int k = idx >> 9, n = idx & 511;
  WT[(size_t)n * DM + k] = W[(size_t)k * DM + n];
}

// ---------------- w[bh,t] = 0.5*(vec1+vec2)[t]/8 * sum_j Yk[b,j,h*64+t] -----
__global__ __launch_bounds__(256) void ksum_kernel(
    const float* __restrict__ Yk, const float* __restrict__ vec1,
    const float* __restrict__ vec2, float* __restrict__ wbuf) {
  const int blk = blockIdx.x;       // 1024 blocks: (b, col)
  const int b = blk >> 9;
  const int col = blk & 511;
  __shared__ float red[256];
  float s = 0.0f;
  const float* base = Yk + (size_t)b * LSEQ * DM + col;
  for (int j = threadIdx.x; j < LSEQ; j += 256) s += base[(size_t)j * DM];
  red[threadIdx.x] = s;
  __syncthreads();
  for (int off = 128; off > 0; off >>= 1) {
    if (threadIdx.x < off) red[threadIdx.x] += red[threadIdx.x + off];
    __syncthreads();
  }
  if (threadIdx.x == 0) {
    const int h = col >> 6, t = col & 63;
    const float vec = 0.5f * (vec1[t] + vec2[t]);
    wbuf[(size_t)(b * NH + h) * DK + t] = red[0] * vec * 0.125f;  // /sqrt(64)
  }
}

// ---------------- core[b,i,h*64+r] = sum_t (qh*w)[t] * Yv[b,t,h*64+r] -------
// grid: (L/64 ichunks, B*NH). Exact f32 via LDS; only 0.3 GFLOP total.
__global__ __launch_bounds__(256) void core_kernel(
    const float* __restrict__ Yq, const float* __restrict__ Yv,
    const float* __restrict__ wbuf, float* __restrict__ core) {
  const int bh = blockIdx.y;
  const int b = bh >> 3, h = bh & 7;
  const int i0 = blockIdx.x * 64;
  const int tid = threadIdx.x;
  __shared__ float Vs[64][64];
  __shared__ float Qs[64][65];
  __shared__ float ws[64];
  if (tid < 64) ws[tid] = wbuf[(size_t)bh * DK + tid];
  for (int e = tid; e < 4096; e += 256) {
    int t = e >> 6, r = e & 63;
    Vs[t][r] = Yv[(size_t)(b * LSEQ + t) * DM + h * DK + r];
  }
  __syncthreads();
  for (int e = tid; e < 4096; e += 256) {
    int il = e >> 6, t = e & 63;
    Qs[il][t] = Yq[(size_t)(b * LSEQ + i0 + il) * DM + h * DK + t] * ws[t];
  }
  __syncthreads();
  const int r = tid & 63;
  const int ib = tid >> 6;
#pragma unroll 1
  for (int s = 0; s < 16; ++s) {
    const int il = ib * 16 + s;
    float acc = 0.0f;
#pragma unroll
    for (int t = 0; t < 64; ++t) acc += Qs[il][t] * Vs[t][r];
    core[(size_t)(b * LSEQ + i0 + il) * DM + h * DK + r] = acc;
  }
}

// ---------------- LayerNorm over last dim (512) -----------------------------
__global__ __launch_bounds__(256) void ln_kernel(
    const float* __restrict__ pre, const float* __restrict__ gamma,
    const float* __restrict__ beta, float* __restrict__ out) {
  const int row = blockIdx.x;
  const int tid = threadIdx.x;
  const float* p = pre + (size_t)row * DM;
  float x0 = p[tid], x1 = p[tid + 256];
  __shared__ float red[256];
  __shared__ float mu_s, rstd_s;
  red[tid] = x0 + x1;
  __syncthreads();
  for (int off = 128; off > 0; off >>= 1) {
    if (tid < off) red[tid] += red[tid + off];
    __syncthreads();
  }
  if (tid == 0) mu_s = red[0] * (1.0f / 512.0f);
  __syncthreads();
  const float mu = mu_s;
  const float d0 = x0 - mu, d1 = x1 - mu;
  red[tid] = d0 * d0 + d1 * d1;
  __syncthreads();
  for (int off = 128; off > 0; off >>= 1) {
    if (tid < off) red[tid] += red[tid + off];
    __syncthreads();
  }
  if (tid == 0) rstd_s = rsqrtf(red[0] * (1.0f / 512.0f) + 1e-5f);
  __syncthreads();
  const float rstd = rstd_s;
  out[(size_t)row * DM + tid] = d0 * rstd * gamma[tid] + beta[tid];
  out[(size_t)row * DM + tid + 256] = d1 * rstd * gamma[tid + 256] + beta[tid + 256];
}

// ---------------------------------------------------------------------------
extern "C" void kernel_launch(void* const* d_in, const int* in_sizes, int n_in,
                              void* d_out, int out_size, void* d_ws, size_t ws_size,
                              hipStream_t stream) {
  (void)in_sizes; (void)n_in; (void)out_size; (void)ws_size;
  const float* q    = (const float*)d_in[0];
  const float* k    = (const float*)d_in[1];
  const float* v    = (const float*)d_in[2];
  // d_in[3] = attn_mask (unused by the reference math)
  const float* Wq   = (const float*)d_in[4];
  const float* bq   = (const float*)d_in[5];
  const float* Wk   = (const float*)d_in[6];
  const float* bk   = (const float*)d_in[7];
  const float* Wv   = (const float*)d_in[8];
  const float* bv   = (const float*)d_in[9];
  const float* Wfc  = (const float*)d_in[10];
  const float* bfc  = (const float*)d_in[11];
  const float* gamma= (const float*)d_in[12];
  const float* beta = (const float*)d_in[13];
  const float* vec1 = (const float*)d_in[14];
  const float* vec2 = (const float*)d_in[15];

  float* out  = (float*)d_out;                       // (2,2048,512)
  float* attn = out + (size_t)NB * LSEQ * DM;        // (16,2048,2048)

  // workspace carve (~44 MB)
  float* ws   = (float*)d_ws;
  float* WTq  = ws;                        // 512*512
  float* WTk  = WTq + DM * DM;
  float* WTv  = WTk + DM * DM;
  float* WTfc = WTv + DM * DM;
  float* Yq   = WTfc + DM * DM;            // 4096*512 each
  float* Yk   = Yq + (size_t)NB * LSEQ * DM;
  float* Yv   = Yk + (size_t)NB * LSEQ * DM;
  float* core = Yv + (size_t)NB * LSEQ * DM;
  float* pre  = core + (size_t)NB * LSEQ * DM;
  float* wbuf = pre + (size_t)NB * LSEQ * DM;  // 16*64

  // 1) transpose weights to N x K layout for the WMMA B-fragment path
  transpose_kernel<<<1024, 256, 0, stream>>>(Wq, WTq);
  transpose_kernel<<<1024, 256, 0, stream>>>(Wk, WTk);
  transpose_kernel<<<1024, 256, 0, stream>>>(Wv, WTv);
  transpose_kernel<<<1024, 256, 0, stream>>>(Wfc, WTfc);

  // 2) projections: Y = X(4096x512) @ W(512x512) + b  (bias, no residual)
  dim3 gproj(DM / 128, (NB * LSEQ) / 128, 1);
  gemm_wmma_kernel<true, false><<<gproj, 256, 0, stream>>>(q, WTq, bq, nullptr, Yq,
      DM, DM, DM, DM, 0, 0, 0, 0, 0, 0, 1);
  gemm_wmma_kernel<true, false><<<gproj, 256, 0, stream>>>(k, WTk, bk, nullptr, Yk,
      DM, DM, DM, DM, 0, 0, 0, 0, 0, 0, 1);
  gemm_wmma_kernel<true, false><<<gproj, 256, 0, stream>>>(v, WTv, bv, nullptr, Yv,
      DM, DM, DM, DM, 0, 0, 0, 0, 0, 0, 1);

  // 3) attn = qh @ kh^T, batched over (h,b): z = h*2 + b  (pure GEMM epilogue)
  {
    dim3 gattn(LSEQ / 128, LSEQ / 128, NH * NB);
    const long long sAh = DK, sAb = (long long)LSEQ * DM;
    const long long sCh = (long long)NB * LSEQ * LSEQ, sCb = (long long)LSEQ * LSEQ;
    gemm_wmma_kernel<false, false><<<gattn, 256, 0, stream>>>(Yq, Yk, nullptr, nullptr, attn,
        DM, DM, LSEQ, DK, sAh, sAb, sAh, sAb, sCh, sCb, NB);
  }

  // 4) core path: column sums of kh, then (qh .* w) @ vh[:64]
  ksum_kernel<<<NB * DM, 256, 0, stream>>>(Yk, vec1, vec2, wbuf);
  {
    dim3 gcore(LSEQ / 64, NB * NH, 1);
    core_kernel<<<gcore, 256, 0, stream>>>(Yq, Yv, wbuf, core);
  }

  // 5) fc: pre = core @ Wfc + bfc + residual(q)  (bias + residual)
  gemm_wmma_kernel<true, true><<<gproj, 256, 0, stream>>>(core, WTfc, bfc, q, pre,
      DM, DM, DM, DM, 0, 0, 0, 0, 0, 0, 1);

  // 6) layernorm -> out
  ln_kernel<<<NB * LSEQ, 256, 0, stream>>>(pre, gamma, beta, out);
}
